// MLP_MoElayer_78812649881949
// MI455X (gfx1250) — matrix operand order repaired and checked by reading, not compiled, not run on hardware
//
#include <hip/hip_runtime.h>
#include <hip/hip_bf16.h>
#include <math.h>

typedef __bf16 bf16;
typedef __attribute__((ext_vector_type(16))) __bf16 v16bf;
typedef __attribute__((ext_vector_type(8)))  __bf16 v8bf;
typedef __attribute__((ext_vector_type(8)))  float  v8f;

#define BATCH 64
#define TOK   1024
#define C_    512
#define E_    5
#define HID_  2048
#define MT    64          // tokens per block
#define XS_STRIDE (C_ + 8)
#define HS_STRIDE (128 + 8)
#define MPARTS 32         // mean reduction parts per batch (32 tokens each)

// ---------- helpers ----------
__device__ __forceinline__ v16bf cat8(v8bf lo, v8bf hi) {
  return __builtin_shufflevector(lo, hi, 0,1,2,3,4,5,6,7,8,9,10,11,12,13,14,15);
}
__device__ __forceinline__ unsigned short bfbits(float f) {
  __bf16 h = (__bf16)f;
  return __builtin_bit_cast(unsigned short, h);
}

// ---------- kernel 1a: partial token sums (2048 blocks -> full HBM concurrency) ----------
__global__ void mean_part_kernel(const float* __restrict__ x, float* __restrict__ partial) {
  int b = blockIdx.x, p = blockIdx.y;            // 32 tokens per part
  int tid = threadIdx.x;
  #pragma unroll
  for (int cc = 0; cc < 2; ++cc) {
    int c = tid + cc * 256;
    const float* ptr = x + ((size_t)b * TOK + (size_t)p * (TOK / MPARTS)) * C_ + c;
    float s = 0.f;
    #pragma unroll 8
    for (int t = 0; t < TOK / MPARTS; ++t) s += ptr[(size_t)t * C_];
    partial[((size_t)b * MPARTS + p) * C_ + c] = s;
  }
}

// ---------- kernel 1b: fold partials -> xg (deterministic fixed-order sum) ----------
__global__ void mean_fold_kernel(const float* __restrict__ partial, float* __restrict__ xg) {
  int idx = blockIdx.x * 256 + threadIdx.x;      // 64*512 = 32768 outputs
  int b = idx >> 9, c = idx & (C_ - 1);
  float s = 0.f;
  #pragma unroll
  for (int p = 0; p < MPARTS; ++p) s += partial[((size_t)b * MPARTS + p) * C_ + c];
  xg[idx] = s * (1.0f / TOK);
}

// ---------- kernel 2: gating (argmax == top-1; softmax of 1 logit == 1.0) ----------
__global__ void gate_kernel(const float* __restrict__ xg, const float* __restrict__ w_gate,
                            int* __restrict__ expert, float* __restrict__ loss_out) {
  __shared__ int cnt[E_];
  int b = threadIdx.x;                           // 64 threads
  if (b < E_) cnt[b] = 0;
  __syncthreads();
  float acc[E_] = {0.f, 0.f, 0.f, 0.f, 0.f};
  for (int c = 0; c < C_; ++c) {
    float xv = xg[b * C_ + c];
    #pragma unroll
    for (int e = 0; e < E_; ++e) acc[e] += xv * w_gate[c * E_ + e];
  }
  int best = 0; float bl = -1e30f;
  #pragma unroll
  for (int e = 0; e < E_; ++e) {
    float l = fminf(50.f, fmaxf(-50.f, acc[e]));
    if (l > bl) { bl = l; best = e; }
  }
  expert[b] = best;
  atomicAdd(&cnt[best], 1);
  __syncthreads();
  if (b == 0) {
    // importance == load == per-expert counts (all gates are exactly 1.0)
    float mean = (float)BATCH / (float)E_;       // 12.8, never near eps
    float var = 0.f;
    #pragma unroll
    for (int e = 0; e < E_; ++e) { float d = (float)cnt[e] - mean; var += d * d; }
    var *= (1.0f / (E_ - 1));                    // ddof=1
    float cv2 = var / (mean * mean + 1e-10f);
    *loss_out = fminf(1000.f, fmaxf(0.f, 2.0f * cv2));
  }
}

// ---------- kernel 3: f32 (E,R,Cc) -> bf16 (E,Cc,R) transpose+convert ----------
__global__ void transpose_cvt(const float* __restrict__ src, bf16* __restrict__ dst,
                              int R, int Cc) {
  __shared__ float tile[32][33];
  int e = blockIdx.z;
  int c0 = blockIdx.x * 32, r0 = blockIdx.y * 32;
  int tx = threadIdx.x, ty = threadIdx.y;        // (32, 8)
  const float* s = src + (size_t)e * R * Cc;
  bf16* d = dst + (size_t)e * R * Cc;
  #pragma unroll
  for (int j = 0; j < 32; j += 8)
    tile[ty + j][tx] = s[(size_t)(r0 + ty + j) * Cc + (c0 + tx)];
  __syncthreads();
  #pragma unroll
  for (int j = 0; j < 32; j += 8)
    d[(size_t)(c0 + ty + j) * R + (r0 + tx)] = (bf16)tile[tx][ty + j];
}

// ---------- kernel 4: fused MoE FFN, bf16 WMMA, f32 accumulate ----------
__global__ __launch_bounds__(256) void ffn_kernel(
    const float* __restrict__ x,
    const bf16*  __restrict__ W1t,   // (E, HID, C)  N-major
    const float* __restrict__ b1,    // (E, HID)
    const bf16*  __restrict__ W2t,   // (E, C, HID)  N-major
    const float* __restrict__ b2,    // (E, C)
    const int*   __restrict__ expert,
    float* __restrict__ y) {
  __shared__ unsigned short Xs[MT * XS_STRIDE];       // 64x512 bf16 tokens
  __shared__ unsigned short Hs[2][MT * HS_STRIDE];    // double-buffered 64x128 bf16 chunk

  const int b = blockIdx.y, t = blockIdx.x;
  const int e = expert[b];
  const int tid  = threadIdx.x;
  const int wave = tid >> 5, lane = tid & 31;
  const int wm = wave & 3, wn = wave >> 2;            // 4 (M) x 2 (N) wave grid
  const int lm = lane & 15, half = lane >> 4;

  // ---- load token tile, convert f32 -> bf16 into LDS (row-major [m][k]) ----
  const float* xb = x + ((size_t)b * TOK + (size_t)t * MT) * C_;
  #pragma unroll
  for (int i = 0; i < 32; ++i) {
    int flat4 = tid + i * 256;                        // MT*C/4 = 8192 float4 groups
    int m = flat4 >> 7;                               // 128 float4 per row
    int k = (flat4 & 127) * 4;
    float4 v = *(const float4*)(xb + (size_t)m * C_ + k);
    unsigned int p0 = (unsigned)bfbits(v.x) | ((unsigned)bfbits(v.y) << 16);
    unsigned int p1 = (unsigned)bfbits(v.z) | ((unsigned)bfbits(v.w) << 16);
    *(uint2*)&Xs[m * XS_STRIDE + k] = make_uint2(p0, p1);
  }
  __syncthreads();

  const bf16*  W1e = W1t + (size_t)e * HID_ * C_;
  const bf16*  W2e = W2t + (size_t)e * C_ * HID_;
  const float* b1e = b1 + e * HID_;
  const float* b2e = b2 + e * C_;

  v8f acc[16];                                        // wave tile: 16(M) x 256(N) of Y
  #pragma unroll
  for (int i = 0; i < 16; ++i) {
    v8f z = {0.f,0.f,0.f,0.f,0.f,0.f,0.f,0.f};
    acc[i] = z;
  }

  for (int hc = 0; hc < HID_ / 128; ++hc) {
    unsigned short* Hbuf = Hs[hc & 1];
    // ---- stage A: H(64x128) = Xs(64x512) @ W1t-chunk^T ----
    v8f hacc[4];
    #pragma unroll
    for (int i = 0; i < 4; ++i) {
      v8f z = {0.f,0.f,0.f,0.f,0.f,0.f,0.f,0.f};
      hacc[i] = z;
    }
    for (int ks = 0; ks < C_; ks += 32) {
      // A fragment: row m=lm, K split per ISA A-layout (half*8, +16 jump)
      const unsigned short* ar = &Xs[(wm * 16 + lm) * XS_STRIDE + ks + half * 8];
      v16bf Af = cat8(*(const v8bf*)ar, *(const v8bf*)(ar + 16));
      #pragma unroll
      for (int nt = 0; nt < 4; ++nt) {
        int n = hc * 128 + wn * 64 + nt * 16 + lm;
        // B fragment: column n, K split per ISA layout (half*8, +16 jump)
        const bf16* br = W1e + (size_t)n * C_ + ks + half * 8;
        v16bf Bf = cat8(*(const v8bf*)br, *(const v8bf*)(br + 16));
        hacc[nt] = __builtin_amdgcn_wmma_f32_16x16x32_bf16(
            false, Af, false, Bf, (short)0, hacc[nt], false, false);
      }
    }
    // ---- bias + exact gelu + convert, store H chunk to LDS ----
    #pragma unroll
    for (int nt = 0; nt < 4; ++nt) {
      float bias = b1e[hc * 128 + wn * 64 + nt * 16 + lm];
      #pragma unroll
      for (int j = 0; j < 8; ++j) {
        float v = hacc[nt][j] + bias;
        v = 0.5f * v * (1.0f + erff(v * 0.70710678118654752f));
        int m = wm * 16 + half * 8 + j;               // C-layout: m = vgpr + 8*half
        Hbuf[m * HS_STRIDE + wn * 64 + nt * 16 + lm] = bfbits(v);
      }
    }
    __syncthreads();  // Hbuf writes -> reads; also (transitively) guards buffer reuse
    // ---- stage B: Y(64x512) += H(64x128) @ W2t-chunk^T ----
    v16bf Ah[4];
    #pragma unroll
    for (int kk = 0; kk < 4; ++kk) {
      const unsigned short* ar = &Hbuf[(wm * 16 + lm) * HS_STRIDE + kk * 32 + half * 8];
      Ah[kk] = cat8(*(const v8bf*)ar, *(const v8bf*)(ar + 16));
    }
    #pragma unroll
    for (int nt = 0; nt < 16; ++nt) {
      int n = wn * 256 + nt * 16 + lm;
      const bf16* base = W2e + (size_t)n * HID_ + hc * 128;
      #pragma unroll
      for (int kk = 0; kk < 4; ++kk) {
        const bf16* br = base + kk * 32 + half * 8;
        v16bf Bf = cat8(*(const v8bf*)br, *(const v8bf*)(br + 16));
        acc[nt] = __builtin_amdgcn_wmma_f32_16x16x32_bf16(
            false, Ah[kk], false, Bf, (short)0, acc[nt], false, false);
      }
    }
    // no trailing barrier: next chunk writes the other Hs buffer
  }

  // ---- epilogue: + b2, write f32 output (gate weight is exactly 1.0) ----
  float* yb = y + ((size_t)b * TOK + (size_t)t * MT) * C_;
  #pragma unroll
  for (int nt = 0; nt < 16; ++nt) {
    int n = wn * 256 + nt * 16 + lm;
    float bias = b2e[n];
    #pragma unroll
    for (int j = 0; j < 8; ++j) {
      int m = wm * 16 + half * 8 + j;
      yb[(size_t)m * C_ + n] = acc[nt][j] + bias;
    }
  }
}

// ---------- host ----------
extern "C" void kernel_launch(void* const* d_in, const int* in_sizes, int n_in,
                              void* d_out, int out_size, void* d_ws, size_t ws_size,
                              hipStream_t stream) {
  (void)in_sizes; (void)n_in; (void)out_size; (void)ws_size;
  const float* x      = (const float*)d_in[0];
  const float* w_gate = (const float*)d_in[1];
  const float* W1     = (const float*)d_in[2];
  const float* b1     = (const float*)d_in[3];
  const float* W2     = (const float*)d_in[4];
  const float* b2     = (const float*)d_in[5];

  float* y = (float*)d_out;
  float* loss_out = y + (size_t)BATCH * TOK * C_;     // tuple: (y, loss)

  char* ws = (char*)d_ws;
  size_t o = 0;
  bf16* W1t = (bf16*)(ws + o); o += (size_t)E_ * HID_ * C_ * sizeof(bf16);
  bf16* W2t = (bf16*)(ws + o); o += (size_t)E_ * C_ * HID_ * sizeof(bf16);
  float* xg = (float*)(ws + o); o += (size_t)BATCH * C_ * sizeof(float);
  int* expert = (int*)(ws + o); o += 64 * sizeof(int);
  float* partial = (float*)(ws + o); o += (size_t)BATCH * MPARTS * C_ * sizeof(float);

  mean_part_kernel<<<dim3(BATCH, MPARTS), 256, 0, stream>>>(x, partial);
  mean_fold_kernel<<<(BATCH * C_) / 256, 256, 0, stream>>>(partial, xg);
  gate_kernel<<<1, 64, 0, stream>>>(xg, w_gate, expert, loss_out);
  // W1 (E, C, HID) -> W1t (E, HID, C)
  transpose_cvt<<<dim3(HID_ / 32, C_ / 32, E_), dim3(32, 8), 0, stream>>>(W1, W1t, C_, HID_);
  // W2 (E, HID, C) -> W2t (E, C, HID)
  transpose_cvt<<<dim3(C_ / 32, HID_ / 32, E_), dim3(32, 8), 0, stream>>>(W2, W2t, HID_, C_);
  ffn_kernel<<<dim3(TOK / MT, BATCH), 256, 0, stream>>>(x, W1t, b1, W2t, b2, expert, y);
}
